// HNet_14800457302192
// MI455X (gfx1250) — compile-verified, matrix operand order checked by reading, not run on hardware
//
#include <hip/hip_runtime.h>
#include <hip/hip_bf16.h>
#include <stdint.h>

// ---------------------------------------------------------------------------
// HNet forward for MI455X (gfx1250, wave32, WMMA + async LDS DMA).
//   K1: fp32 -> bf16 conversion of hidden_states / Wq / Wk
//   K2: router GEMM: v_wmma_f32_16x16x32_bf16 with weights staged into LDS
//       via global_load_async_to_lds_b128 (ASYNCcnt, double-buffered);
//       per-row qq/kk/qk folded in-register + lane butterflies -> prob[b,l]
//   K3/K4/K5: segmented linear-recurrence scan (EMA) + residual add
// ---------------------------------------------------------------------------

typedef __attribute__((ext_vector_type(16))) __bf16 v16bf;
typedef __attribute__((ext_vector_type(8)))  float  v8f;

constexpr int B_ = 4, L_ = 4096, D_ = 1024;
constexpr int SEGS = 64, SEGLEN = L_ / SEGS;   // 64 tokens / segment
constexpr int TILES = L_ / 16;                 // 256 16-token tiles per batch

// LDS staging geometry: per chunk, 2 matrices x 64 rows x 128 bf16 columns.
// Row stride padded to 136 elements (272 B -> bank rotation of 4/row).
constexpr int ROWE   = 136;                    // elements per staged row
constexpr int MATE   = 64 * ROWE;              // 8704 elements per matrix
constexpr int BUFE   = 2 * MATE;               // 17408 elements per buffer
constexpr int SMEME  = 2 * BUFE;               // two buffers (69632 B)

__device__ __forceinline__ uint16_t f32_to_bf16(float f) {
  uint32_t x = __float_as_uint(f);
  uint32_t r = x + 0x7fffu + ((x >> 16) & 1u);   // round-to-nearest-even
  return (uint16_t)(r >> 16);
}

// ------------------------------- K1: convert -------------------------------
__global__ void cvt_bf16_kernel(const float* __restrict__ src,
                                uint16_t* __restrict__ dst, int n) {
  int i = blockIdx.x * blockDim.x + threadIdx.x;
  int stride = gridDim.x * blockDim.x;
  for (; i < n; i += stride) dst[i] = f32_to_bf16(src[i]);
}

// Issue one 32KB weight chunk (2 mats x 64 rows x 256B) as 8 async-b128
// instructions per wave; each instruction moves 16B per lane (512B/wave).
__device__ __forceinline__ void stage_chunk(uint16_t* smem, int buf, int ng,
                                            int chunk,
                                            const uint16_t* __restrict__ wqb,
                                            const uint16_t* __restrict__ wkb,
                                            int tid) {
#pragma unroll
  for (int i = 0; i < 8; ++i) {
    const int idx = tid + 256 * i;    // 0..2047 16-byte transfer id
    const int mat = idx >> 10;        // 0 = Wq, 1 = Wk
    const int rem = idx & 1023;
    const int r   = rem >> 4;         // staged row 0..63
    const int c16 = rem & 15;         // 16B unit within the 256B row
    const uint16_t* src = (mat ? wkb : wqb) +
        ((size_t)(ng * 64 + r) * D_ + chunk * 128 + c16 * 8);
    const uint32_t dst = (uint32_t)(uintptr_t)
        &smem[(size_t)buf * BUFE + mat * MATE + r * ROWE + c16 * 8];
    asm volatile("global_load_async_to_lds_b128 %0, %1, off"
                 :: "v"(dst), "v"(src) : "memory");
  }
}

// ------------------------------- K2: router --------------------------------
__global__ void __launch_bounds__(256)
router_kernel(const uint16_t* __restrict__ hsb,
              const uint16_t* __restrict__ wqb,
              const uint16_t* __restrict__ wkb,
              float* __restrict__ prob) {
  __shared__ uint16_t smem[SMEME];

  const int tid  = threadIdx.x;
  const int lane = tid & 31;
  const int wav  = tid >> 5;
  const int gt   = blockIdx.x * 8 + wav;   // 0..B*TILES-1
  const int b    = gt >> 8;                // / TILES
  const int tile = gt & (TILES - 1);
  const int t0   = tile * 16;
  const int h    = lane >> 4;              // lane half
  const int col  = lane & 15;

  if (tile == 0 && lane == 0) prob[(size_t)b * L_] = 1.0f;  // prob[:,0] = 1

  // bf16-pair K offsets (A and B fragments share the pattern):
  // VGPR v holds K = {0,2,4,6,16,18,20,22}[v] + 8*half, pair (k, k+1).
  int koff[8];
#pragma unroll
  for (int v = 0; v < 8; ++v)
    koff[v] = ((v < 4) ? 2 * v : 16 + 2 * (v - 4)) + 8 * h;

  const int tq = t0 + col;                 // q row (A fragment row m)
  const int tk = (t0 + col + 1 < L_) ? t0 + col + 1 : L_ - 1;
  const uint16_t* rowq = hsb + ((size_t)b * L_ + tq) * D_;
  const uint16_t* rowk = hsb + ((size_t)b * L_ + tk) * D_;

  float qq[8], kk[8], qk[8];
#pragma unroll
  for (int v = 0; v < 8; ++v) { qq[v] = 0.f; kk[v] = 0.f; qk[v] = 0.f; }

  const v8f zero = {0.f, 0.f, 0.f, 0.f, 0.f, 0.f, 0.f, 0.f};

#pragma unroll 1
  for (int ng = 0; ng < 16; ++ng) {        // 64 weight rows per group
    v8f accq[4], acck[4];
#pragma unroll
    for (int j = 0; j < 4; ++j) { accq[j] = zero; acck[j] = zero; }

    stage_chunk(smem, 0, ng, 0, wqb, wkb, tid);   // preload chunk 0

#pragma unroll 1
    for (int c = 0; c < 8; ++c) {          // 8 K-chunks of 128 columns
      if (c < 7) {
        stage_chunk(smem, (c + 1) & 1, ng, c + 1, wqb, wkb, tid);
        asm volatile("s_wait_asynccnt 0x8" ::: "memory");  // chunk c done
      } else {
        asm volatile("s_wait_asynccnt 0x0" ::: "memory");
      }
      __syncthreads();                     // all waves' copies visible

      const uint16_t* base = &smem[(size_t)(c & 1) * BUFE];
#pragma unroll
      for (int ktl = 0; ktl < 4; ++ktl) {  // 4 x K32 per chunk
        const int kb = c * 128 + ktl * 32;
        union { v16bf v; uint32_t u[8]; } aq, ak;
#pragma unroll
        for (int v = 0; v < 8; ++v) {
          aq.u[v] = *reinterpret_cast<const uint32_t*>(rowq + kb + koff[v]);
          ak.u[v] = *reinterpret_cast<const uint32_t*>(rowk + kb + koff[v]);
        }
#pragma unroll
        for (int j = 0; j < 4; ++j) {
          const int rloc = j * 16 + col;   // row within the staged 64
          const uint16_t* lq = base + rloc * ROWE + ktl * 32;
          const uint16_t* lk = lq + MATE;
          union { v16bf v; uint32_t u[8]; } bq, bk;
#pragma unroll
          for (int v = 0; v < 8; ++v) {
            bq.u[v] = *reinterpret_cast<const uint32_t*>(lq + koff[v]);
            bk.u[v] = *reinterpret_cast<const uint32_t*>(lk + koff[v]);
          }
          accq[j] = __builtin_amdgcn_wmma_f32_16x16x32_bf16(
              false, aq.v, false, bq.v, (short)0, accq[j], false, false);
          acck[j] = __builtin_amdgcn_wmma_f32_16x16x32_bf16(
              false, ak.v, false, bk.v, (short)0, acck[j], false, false);
        }
      }
      __syncthreads();                     // buffer may be overwritten next
    }

    // Fold 4 finished C tiles into per-row partial sums over the N dim.
#pragma unroll
    for (int j = 0; j < 4; ++j)
#pragma unroll
      for (int v = 0; v < 8; ++v) {
        float q = accq[j][v], k = acck[j][v];
        qq[v] += q * q; kk[v] += k * k; qk[v] += q * k;
      }
  }

  // Finish the N reduction across the 16 lanes of each half-wave.
#pragma unroll
  for (int v = 0; v < 8; ++v) {
#pragma unroll
    for (int m = 1; m <= 8; m <<= 1) {
      qq[v] += __shfl_xor(qq[v], m, 32);
      kk[v] += __shfl_xor(kk[v], m, 32);
      qk[v] += __shfl_xor(qk[v], m, 32);
    }
  }

  // Row m = v + 8*h; lane with col==v writes it (static register indices).
#pragma unroll
  for (int v = 0; v < 8; ++v) {
    if (col == v) {
      const int t = t0 + v + 8 * h;        // q token index l
      if (t <= L_ - 2) {
        float nq = fmaxf(sqrtf(qq[v]), 1e-12f);
        float nk = fmaxf(sqrtf(kk[v]), 1e-12f);
        float cs = qk[v] / (nq * nk);
        float p  = 0.5f * (1.0f - cs);
        p = fminf(fmaxf(p, 0.0f), 1.0f);
        prob[(size_t)b * L_ + t + 1] = p;
      }
    }
  }
}

// ------------------------ K3: per-segment affine fold ----------------------
__global__ void __launch_bounds__(256)
seg_reduce_kernel(const float* __restrict__ hs, const float* __restrict__ prob,
                  float* __restrict__ segA, float* __restrict__ segB) {
  const int b  = blockIdx.x / SEGS;
  const int s  = blockIdx.x % SEGS;
  const int d0 = threadIdx.x * 4;
  float A = 1.0f;
  float4 Bacc = make_float4(0.f, 0.f, 0.f, 0.f);
  const int tbeg = s * SEGLEN;
  for (int t = tbeg; t < tbeg + SEGLEN; ++t) {
    const size_t row = ((size_t)b * L_ + t) * D_ + d0;
    __builtin_prefetch(hs + row + D_, 0, 0);     // global_prefetch_b8
    float p = prob[(size_t)b * L_ + t];
    if (p > 0.5f) {
      float a = 1.0f - p;
      const float4 x = *reinterpret_cast<const float4*>(hs + row);
      A *= a;
      Bacc.x = a * Bacc.x + p * x.x;
      Bacc.y = a * Bacc.y + p * x.y;
      Bacc.z = a * Bacc.z + p * x.z;
      Bacc.w = a * Bacc.w + p * x.w;
    }
  }
  *reinterpret_cast<float4*>(segB + (size_t)(b * SEGS + s) * D_ + d0) = Bacc;
  if (threadIdx.x == 0) segA[b * SEGS + s] = A;
}

// --------------------- K4: scan over the 64 segments -----------------------
__global__ void __launch_bounds__(256)
seg_scan_kernel(const float* __restrict__ init_state,
                const float* __restrict__ segA, const float* __restrict__ segB,
                float* __restrict__ inst) {
  const int b  = blockIdx.x;
  const int d0 = threadIdx.x * 4;
  float4 st = *reinterpret_cast<const float4*>(init_state + (size_t)b * D_ + d0);
  for (int s = 0; s < SEGS; ++s) {
    *reinterpret_cast<float4*>(inst + (size_t)(b * SEGS + s) * D_ + d0) = st;
    float a = segA[b * SEGS + s];
    const float4 bb =
        *reinterpret_cast<const float4*>(segB + (size_t)(b * SEGS + s) * D_ + d0);
    st.x = a * st.x + bb.x;
    st.y = a * st.y + bb.y;
    st.z = a * st.z + bb.z;
    st.w = a * st.w + bb.w;
  }
}

// ------------------- K5: apply scan + residual add -------------------------
__global__ void __launch_bounds__(256)
ema_apply_kernel(const float* __restrict__ hs, const float* __restrict__ residual,
                 const float* __restrict__ prob, const float* __restrict__ inst,
                 float* __restrict__ out) {
  const int b  = blockIdx.x / SEGS;
  const int s  = blockIdx.x % SEGS;
  const int d0 = threadIdx.x * 4;
  float4 ema = *reinterpret_cast<const float4*>(inst + (size_t)(b * SEGS + s) * D_ + d0);
  const int tbeg = s * SEGLEN;
  for (int t = tbeg; t < tbeg + SEGLEN; ++t) {
    const size_t row = ((size_t)b * L_ + t) * D_ + d0;
    __builtin_prefetch(residual + row + D_, 0, 0);
    float p = prob[(size_t)b * L_ + t];
    if (p > 0.5f) {
      float a = 1.0f - p;
      const float4 x = *reinterpret_cast<const float4*>(hs + row);
      ema.x = a * ema.x + p * x.x;
      ema.y = a * ema.y + p * x.y;
      ema.z = a * ema.z + p * x.z;
      ema.w = a * ema.w + p * x.w;
    }
    const float4 r = *reinterpret_cast<const float4*>(residual + row);
    float4 o;
    o.x = r.x + ema.x; o.y = r.y + ema.y; o.z = r.z + ema.z; o.w = r.w + ema.w;
    *reinterpret_cast<float4*>(out + row) = o;
  }
}

// ------------------------------- host glue ---------------------------------
extern "C" void kernel_launch(void* const* d_in, const int* in_sizes, int n_in,
                              void* d_out, int out_size, void* d_ws, size_t ws_size,
                              hipStream_t stream) {
  const float* hs  = (const float*)d_in[0];   // (B, L, D)
  const float* res = (const float*)d_in[1];   // (B, L, D)
  const float* wq  = (const float*)d_in[2];   // (D, D)
  const float* wk  = (const float*)d_in[3];   // (D, D)
  const float* ini = (const float*)d_in[4];   // (B, D)
  float* out = (float*)d_out;

  char*  ws = (char*)d_ws;
  size_t off = 0;
  auto alloc = [&](size_t bytes) -> void* {
    void* p = ws + off;
    off = (off + bytes + 255) & ~(size_t)255;
    return p;
  };
  uint16_t* hsb  = (uint16_t*)alloc((size_t)B_ * L_ * D_ * 2);
  uint16_t* wqb  = (uint16_t*)alloc((size_t)D_ * D_ * 2);
  uint16_t* wkb  = (uint16_t*)alloc((size_t)D_ * D_ * 2);
  float*    prob = (float*)   alloc((size_t)B_ * L_ * 4);
  float*    segA = (float*)   alloc((size_t)B_ * SEGS * 4);
  float*    segB = (float*)   alloc((size_t)B_ * SEGS * D_ * 4);
  float*    inst = (float*)   alloc((size_t)B_ * SEGS * D_ * 4);

  // K1: bf16 conversions
  cvt_bf16_kernel<<<2048, 256, 0, stream>>>(hs, hsb, B_ * L_ * D_);
  cvt_bf16_kernel<<<1024, 256, 0, stream>>>(wq, wqb, D_ * D_);
  cvt_bf16_kernel<<<1024, 256, 0, stream>>>(wk, wkb, D_ * D_);

  // K2: router — B*TILES = 1024 waves, 8 waves per block share LDS weights
  router_kernel<<<(B_ * TILES) / 8, 256, 0, stream>>>(hsb, wqb, wkb, prob);

  // K3..K5: segmented EMA scan + output
  seg_reduce_kernel<<<B_ * SEGS, 256, 0, stream>>>(hs, prob, segA, segB);
  seg_scan_kernel<<<B_, 256, 0, stream>>>(ini, segA, segB, inst);
  ema_apply_kernel<<<B_ * SEGS, 256, 0, stream>>>(hs, res, prob, inst, out);
}
